// NStransformer_39805756899722
// MI455X (gfx1250) — matrix operand order repaired and checked by reading, not compiled, not run on hardware
//
#include <hip/hip_runtime.h>
#include <math.h>

// ---------------------------------------------------------------------------
// CDNA5 (gfx1250) wave32 WMMA transformer forward, v4.
// All GEMM operands pre-packed to f16 (weights in LDS-panel layout, V
// pre-transposed per head), so fragment feeds are pure 16B loads and LDS
// staging uses GLOBAL_LOAD_ASYNC_TO_LDS_B128 (ASYNCcnt) when available.
// ---------------------------------------------------------------------------

typedef __attribute__((ext_vector_type(16))) _Float16 v16h;
typedef __attribute__((ext_vector_type(8)))  _Float16 v8h;
typedef __attribute__((ext_vector_type(8)))  float    v8f;

#define D_MODEL 512
#define D_FF    2048
#define N_HEADS 8
#define HEAD_E  64
#define BATCH   2
#define SEQ     2048
#define ROWS    (BATCH * SEQ)   // 4096

// ---- async global->LDS copy (16B granule), with sync fallback --------------
#if defined(__gfx1250__) && __has_builtin(__builtin_amdgcn_global_load_async_to_lds_b128)
#define AS1 __attribute__((address_space(1)))
#define AS3 __attribute__((address_space(3)))
typedef int v4i_b __attribute__((vector_size(16)));   // builtin's V4i pointee
__device__ __forceinline__ void async_cp16(_Float16* lds, const _Float16* g) {
    __builtin_amdgcn_global_load_async_to_lds_b128((AS1 v4i_b*)g, (AS3 v4i_b*)lds, 0, 0);
}
__device__ __forceinline__ void async_wait() {
#if __has_builtin(__builtin_amdgcn_s_wait_asynccnt)
    __builtin_amdgcn_s_wait_asynccnt(0);
#else
    asm volatile("s_wait_asynccnt 0x0" ::: "memory");
#endif
}
#else
__device__ __forceinline__ void async_cp16(_Float16* lds, const _Float16* g) {
    *(v8h*)lds = *(const v8h*)g;
}
__device__ __forceinline__ void async_wait() {}
#endif

__device__ __forceinline__ float gelu_f(float x) {
    float x3 = x * x * x;
    return 0.5f * x * (1.0f + tanhf(0.7978845608028654f * (x + 0.044715f * x3)));
}

// 16x32 f16 A-fragment from two aligned 16B runs (ISA 7.12.2):
// af[0..7] <- K = 8*hl+0..7 ; af[8..15] <- K = 16+8*hl+0..7
__device__ __forceinline__ v16h a_frag16(const _Float16* p0, const _Float16* p1) {
    v8h lo = *(const v8h*)p0;
    v8h hi = *(const v8h*)p1;
    return __builtin_shufflevector(lo, hi, 0,1,2,3,4,5,6,7,8,9,10,11,12,13,14,15);
}

// ---------------------------------------------------------------------------
// Pack f32 weights [K][N] -> f16 panels [K/32][N][32]  (LDS-ready layout)
// ---------------------------------------------------------------------------
__global__ __launch_bounds__(256)
void pack_w_k(const float* __restrict__ W, _Float16* __restrict__ P, int K, int N)
{
    int idx = blockIdx.x * 256 + threadIdx.x;   // consecutive n -> coalesced read
    int k = idx / N, n = idx % N;
    P[(size_t)(k >> 5) * N * 32 + (size_t)n * 32 + (k & 31)] = (_Float16)W[idx];
}

__global__ __launch_bounds__(256)
void f32_to_f16_k(const float* __restrict__ X, _Float16* __restrict__ Y)
{
    int idx = blockIdx.x * 256 + threadIdx.x;
    Y[idx] = (_Float16)X[idx];
}

// ---------------------------------------------------------------------------
// GEMM: 128 threads = 4 waves, block tile 64x64, wave = 16 rows x 64 cols
// (4 WMMAs per K-step).  A is f16 row-major.  Wp is f16 panel layout, staged
// to LDS with async b128 copies.  Output modes:
//   omode 0: f32 row-major to Cf     omode 1: f16 row-major to Ch
//   omode 2: f16 per-head transpose [b][h][e][s] to Ch (for V)
// ---------------------------------------------------------------------------
__global__ __launch_bounds__(128)
void gemm_wmma(const _Float16* __restrict__ A, const _Float16* __restrict__ Wp,
               const float* __restrict__ bias, const float* __restrict__ R,
               float* __restrict__ Cf, _Float16* __restrict__ Ch,
               int M, int N, int K, int act, int omode)
{
    __shared__ __attribute__((aligned(32))) _Float16 WtT[64 * 32];  // [n][k]
    const int tid  = threadIdx.x;
    const int wave = tid >> 5;
    const int lane = tid & 31;
    const int hl   = lane >> 4;
    const int ln   = lane & 15;
    const int my   = blockIdx.y * 64 + wave * 16;
    const int nx0  = blockIdx.x * 64;
    const size_t arow = (size_t)(my + ln) * K;

    v8f acc[4];
    v8f vzero = {};
#pragma unroll
    for (int j = 0; j < 4; ++j) acc[j] = vzero;

    for (int k0 = 0; k0 < K; k0 += 32) {
        // contiguous 4KB panel: Wp + panel*N*32 + nx0*32
        const _Float16* gp = Wp + (size_t)(k0 >> 5) * N * 32 + (size_t)nx0 * 32;
#pragma unroll
        for (int i = 0; i < 2; ++i) {
            int c = tid + 128 * i;                 // 256 x 16B chunks
            async_cp16(&WtT[c * 8], gp + c * 8);
        }
        if (k0 + 32 < K) __builtin_prefetch(gp + 2048, 0, 1);  // next panel
        async_wait();
        __syncthreads();

        v16h af = a_frag16(&A[arow + k0 + 8 * hl], &A[arow + k0 + 16 + 8 * hl]);
#pragma unroll
        for (int j = 0; j < 4; ++j) {
            v16h bf = a_frag16(&WtT[(16 * j + ln) * 32 + 8 * hl], &WtT[(16 * j + ln) * 32 + 16 + 8 * hl]);
            acc[j] = __builtin_amdgcn_wmma_f32_16x16x32_f16(false, af, false, bf,
                                                            (short)0, acc[j],
                                                            false, false);
        }
        __syncthreads();
    }

#pragma unroll
    for (int j = 0; j < 4; ++j)
#pragma unroll
        for (int r = 0; r < 8; ++r) {              // C: row = 8*hl + r, col = lane%16
            int row = my + 8 * hl + r;
            int col = nx0 + 16 * j + ln;
            float vv = acc[j][r] + bias[col];
            if (R)   vv += R[(size_t)row * N + col];
            if (act) vv = gelu_f(vv);
            if (omode == 0) {
                Cf[(size_t)row * N + col] = vv;
            } else if (omode == 1) {
                Ch[(size_t)row * N + col] = (_Float16)vv;
            } else {                               // V transpose [b][h][e][s]
                int bb = row >> 11, s = row & (SEQ - 1);
                int hh = col >> 6,  e = col & 63;
                Ch[(((size_t)bb * N_HEADS + hh) * HEAD_E + e) * SEQ + s] = (_Float16)vv;
            }
        }
}

// ---------------------------------------------------------------------------
// Flash attention: 128 threads = 4 waves per (b,h, 64 query rows).
// Q/K f16 row-major, V f16 pre-transposed [b][h][e][s]; K/V tiles staged with
// async b128 copies (no conversion, no VGPR round-trip).  Output f16.
// ---------------------------------------------------------------------------
__global__ __launch_bounds__(128)
void attn_wmma(const _Float16* __restrict__ Qh, const _Float16* __restrict__ Kh,
               const _Float16* __restrict__ Vt, const float* __restrict__ tau,
               const float* __restrict__ delta, _Float16* __restrict__ Oh)
{
    __shared__ __attribute__((aligned(32))) _Float16 Kt [32 * 64];  // [s][e]
    __shared__ __attribute__((aligned(32))) _Float16 VtT[64 * 32];  // [e][s]
    __shared__ __attribute__((aligned(32))) _Float16 Pt [4 * 16 * 32];

    const int tid  = threadIdx.x;
    const int wave = tid >> 5;
    const int lane = tid & 31;
    const int hl   = lane >> 4;
    const int ln   = lane & 15;
    const int m0   = blockIdx.x * 64 + wave * 16;
    const int b    = blockIdx.y / N_HEADS;
    const int h    = blockIdx.y % N_HEADS;
    const float tauv  = tau[b];
    const float scale = 0.125f;                    // 1/sqrt(64)
    _Float16* Pw = Pt + wave * (16 * 32);

    const _Float16* kbase = Kh + ((size_t)b * SEQ) * D_MODEL + h * HEAD_E;
    const _Float16* vbase = Vt + (((size_t)b * N_HEADS + h) * HEAD_E) * SEQ;

    // Preload Q fragments: pure 16B loads
    v16h qa[2];
    {
        const _Float16* qrow = Qh + ((size_t)(b * SEQ + m0 + ln)) * D_MODEL + h * HEAD_E;
#pragma unroll
        for (int c = 0; c < 2; ++c)
            qa[c] = a_frag16(qrow + 32 * c + 8 * hl, qrow + 32 * c + 16 + 8 * hl);
    }

    float mrun[8], lrun[8];
    v8f   oacc[4];
    v8f   vzero = {};
#pragma unroll
    for (int r = 0; r < 8; ++r) { mrun[r] = -3.0e38f; lrun[r] = 0.0f; }
#pragma unroll
    for (int ec = 0; ec < 4; ++ec) oacc[ec] = vzero;

    const int send = blockIdx.x * 64 + 63;         // uniform across block
    for (int s0 = 0; s0 <= send; s0 += 32) {
        // stage K tile [s][e] (32x64) and V tile [e][s] (64x32): 256+256 chunks
#pragma unroll
        for (int i = 0; i < 2; ++i) {
            int c = tid + 128 * i;
            int kr = c >> 3, ke = c & 7;           // K: row kr, 8-half chunk ke
            async_cp16(&Kt[kr * 64 + ke * 8],
                       kbase + (size_t)(s0 + kr) * D_MODEL + ke * 8);
            int ve = c >> 2, vs = c & 3;           // V: e-row ve, chunk vs
            async_cp16(&VtT[ve * 32 + vs * 8],
                       vbase + (size_t)ve * SEQ + s0 + vs * 8);
        }
        async_wait();
        __syncthreads();

        // S tiles: two 16-col halves; K^T fragment contiguous in [s][e]
        v8f S[2];
#pragma unroll
        for (int sc = 0; sc < 2; ++sc) {
            v8f a = vzero;
#pragma unroll
            for (int c = 0; c < 2; ++c) {
                v16h bf = a_frag16(&Kt[(16 * sc + ln) * 64 + 32 * c + 8 * hl], &Kt[(16 * sc + ln) * 64 + 32 * c + 16 + 8 * hl]);
                a = __builtin_amdgcn_wmma_f32_16x16x32_f16(false, qa[c], false, bf,
                                                           (short)0, a, false, false);
            }
            S[sc] = a;
        }

        // online softmax (row r lives in this half-wave's 16-lane group)
        float p0v[8], p1v[8], corr[8];
#pragma unroll
        for (int r = 0; r < 8; ++r) {
            int gm = m0 + 8 * hl + r;
            int sA = s0 + ln, sB = s0 + 16 + ln;
            float z0 = scale * (S[0][r] * tauv + delta[b * SEQ + sA]);
            float z1 = scale * (S[1][r] * tauv + delta[b * SEQ + sB]);
            if (sA > gm) z0 = -3.0e38f;            // causal mask
            if (sB > gm) z1 = -3.0e38f;
            float zm = fmaxf(z0, z1);
            for (int d = 1; d < 16; d <<= 1) zm = fmaxf(zm, __shfl_xor(zm, d, 32));
            float mnew = fmaxf(mrun[r], zm);
            float p0 = __expf(z0 - mnew), p1 = __expf(z1 - mnew);
            float ps = p0 + p1;
            for (int d = 1; d < 16; d <<= 1) ps += __shfl_xor(ps, d, 32);
            float cf = __expf(mrun[r] - mnew);
            lrun[r] = lrun[r] * cf + ps;
            mrun[r] = mnew;
            corr[r] = cf; p0v[r] = p0; p1v[r] = p1;
        }
#pragma unroll
        for (int ec = 0; ec < 4; ++ec)
#pragma unroll
            for (int r = 0; r < 8; ++r) oacc[ec][r] *= corr[r];

        // bounce P through LDS: C-layout -> A-layout (per-wave region)
        __syncthreads();
#pragma unroll
        for (int r = 0; r < 8; ++r) {
            Pw[(8 * hl + r) * 32 + ln]      = (_Float16)p0v[r];
            Pw[(8 * hl + r) * 32 + 16 + ln] = (_Float16)p1v[r];
        }
        __syncthreads();

        v16h pf = a_frag16(&Pw[ln * 32 + 8 * hl], &Pw[ln * 32 + 16 + 8 * hl]);

        // O(16x64) += P(16x32) x V(32x64); V fragment contiguous in [e][s]
#pragma unroll
        for (int ec = 0; ec < 4; ++ec) {
            v16h vf = a_frag16(&VtT[(16 * ec + ln) * 32 + 8 * hl], &VtT[(16 * ec + ln) * 32 + 16 + 8 * hl]);
            oacc[ec] = __builtin_amdgcn_wmma_f32_16x16x32_f16(false, pf, false, vf,
                                                              (short)0, oacc[ec],
                                                              false, false);
        }
        __syncthreads();
    }

#pragma unroll
    for (int ec = 0; ec < 4; ++ec)
#pragma unroll
        for (int r = 0; r < 8; ++r) {
            int gm = m0 + 8 * hl + r;
            Oh[((size_t)(b * SEQ + gm)) * D_MODEL + h * HEAD_E + 16 * ec + ln] =
                (_Float16)(oacc[ec][r] / lrun[r]);
        }
}

// ---------------------------------------------------------------------------
// Row LayerNorm over D=512; optional dual f32 + f16 output.  In-place safe.
// ---------------------------------------------------------------------------
__global__ __launch_bounds__(256)
void layernorm_k(const float* __restrict__ X, const float* __restrict__ g,
                 const float* __restrict__ bta, float* __restrict__ Y,
                 _Float16* __restrict__ Yh)
{
    __shared__ float red[256];
    const int row = blockIdx.x, tid = threadIdx.x;
    const float x0 = X[(size_t)row * 512 + tid];
    const float x1 = X[(size_t)row * 512 + 256 + tid];

    red[tid] = x0 + x1;
    __syncthreads();
    for (int s = 128; s > 0; s >>= 1) { if (tid < s) red[tid] += red[tid + s]; __syncthreads(); }
    float mean = red[0] * (1.0f / 512.0f);
    __syncthreads();

    float d0 = x0 - mean, d1 = x1 - mean;
    red[tid] = d0 * d0 + d1 * d1;
    __syncthreads();
    for (int s = 128; s > 0; s >>= 1) { if (tid < s) red[tid] += red[tid + s]; __syncthreads(); }
    float inv = rsqrtf(red[0] * (1.0f / 512.0f) + 1e-5f);

    float y0 = d0 * inv * g[tid]       + bta[tid];
    float y1 = d1 * inv * g[tid + 256] + bta[tid + 256];
    Y[(size_t)row * 512 + tid]       = y0;
    Y[(size_t)row * 512 + 256 + tid] = y1;
    if (Yh) {
        Yh[(size_t)row * 512 + tid]       = (_Float16)y0;
        Yh[(size_t)row * 512 + 256 + tid] = (_Float16)y1;
    }
}

// ---------------------------------------------------------------------------
extern "C" void kernel_launch(void* const* d_in, const int* in_sizes, int n_in,
                              void* d_out, int out_size, void* d_ws, size_t ws_size,
                              hipStream_t stream)
{
    (void)in_sizes; (void)n_in; (void)out_size; (void)ws_size;

    const float* x_in   = (const float*)d_in[0];
    const float* tau    = (const float*)d_in[1];
    const float* delta  = (const float*)d_in[2];
    /* d_in[3] = att_mask (causal triu) -- applied analytically */
    const float* Wq = (const float*)d_in[4];  const float* bq = (const float*)d_in[5];
    const float* Wk = (const float*)d_in[6];  const float* bk = (const float*)d_in[7];
    const float* Wv = (const float*)d_in[8];  const float* bv = (const float*)d_in[9];
    const float* Wo = (const float*)d_in[10]; const float* bo = (const float*)d_in[11];
    const float* W1 = (const float*)d_in[12]; const float* b1 = (const float*)d_in[13];
    const float* W2 = (const float*)d_in[14]; const float* b2 = (const float*)d_in[15];
    const float* ln1g = (const float*)d_in[16]; const float* ln1b = (const float*)d_in[17];
    const float* ln2g = (const float*)d_in[18]; const float* ln2b = (const float*)d_in[19];
    const float* lnfg = (const float*)d_in[20]; const float* lnfb = (const float*)d_in[21];
    float* out = (float*)d_out;

    // ---- workspace layout -------------------------------------------------
    const size_t NA = (size_t)ROWS * D_MODEL;       // 2M elements
    char* wsb = (char*)d_ws;
    float*    xcur = (float*)wsb;                           wsb += NA * 4;  // 8MB
    float*    xb   = (float*)wsb;                           wsb += NA * 4;  // 8MB
    _Float16* qh   = (_Float16*)wsb;                        wsb += NA * 2;  // 4MB
    _Float16* kh   = (_Float16*)wsb;                        wsb += NA * 2;  // 4MB
    _Float16* vtT  = (_Float16*)wsb;                        wsb += NA * 2;  // 4MB
    _Float16* abh  = (_Float16*)wsb;                        wsb += NA * 2;  // 4MB
    _Float16* hbh  = qh;                   // FFN hidden [4096,2048] f16 overlays
                                           // dead qh..abh (exactly 8M halfs)
    _Float16* xh   = (_Float16*)wsb;                        wsb += NA * 2;  // 4MB
    _Float16* xbh  = (_Float16*)wsb;                        wsb += NA * 2;  // 4MB
    _Float16* wp   = (_Float16*)wsb;       // packed weights: 6M halfs, 12MB
    const size_t WSM = (size_t)D_MODEL * D_MODEL;   // 256K
    const size_t WSF = (size_t)D_MODEL * D_FF;      // 1M
    // per-layer packed pointers: [wq wk wv wo w1 w2] x 2 layers
    _Float16* wp_l[2][6];
    {
        _Float16* p = wp;
        for (int l = 0; l < 2; ++l) {
            for (int i = 0; i < 4; ++i) { wp_l[l][i] = p; p += WSM; }
            wp_l[l][4] = p; p += WSF;
            wp_l[l][5] = p; p += WSF;
        }
    }

    // ---- pack weights to f16 panel layout --------------------------------
    for (int l = 0; l < 2; ++l) {
        pack_w_k<<<(int)(WSM / 256), 256, 0, stream>>>(Wq + l * WSM, wp_l[l][0], D_MODEL, D_MODEL);
        pack_w_k<<<(int)(WSM / 256), 256, 0, stream>>>(Wk + l * WSM, wp_l[l][1], D_MODEL, D_MODEL);
        pack_w_k<<<(int)(WSM / 256), 256, 0, stream>>>(Wv + l * WSM, wp_l[l][2], D_MODEL, D_MODEL);
        pack_w_k<<<(int)(WSM / 256), 256, 0, stream>>>(Wo + l * WSM, wp_l[l][3], D_MODEL, D_MODEL);
        pack_w_k<<<(int)(WSF / 256), 256, 0, stream>>>(W1 + l * WSF, wp_l[l][4], D_MODEL, D_FF);
        pack_w_k<<<(int)(WSF / 256), 256, 0, stream>>>(W2 + l * WSF, wp_l[l][5], D_FF, D_MODEL);
    }

    (void)hipMemcpyAsync(xcur, x_in, NA * sizeof(float), hipMemcpyDeviceToDevice, stream);
    f32_to_f16_k<<<(int)(NA / 256), 256, 0, stream>>>(x_in, xh);

    dim3 blk(128);
    dim3 g512(D_MODEL / 64, ROWS / 64);             // 8 x 64
    dim3 g2048(D_FF / 64, ROWS / 64);               // 32 x 64
    dim3 agrid(SEQ / 64, BATCH * N_HEADS);          // 32 x 16

    for (int l = 0; l < 2; ++l) {
        gemm_wmma<<<g512, blk, 0, stream>>>(xh, wp_l[l][0], bq + l * D_MODEL, nullptr,
                                            nullptr, qh, ROWS, D_MODEL, D_MODEL, 0, 1);
        gemm_wmma<<<g512, blk, 0, stream>>>(xh, wp_l[l][1], bk + l * D_MODEL, nullptr,
                                            nullptr, kh, ROWS, D_MODEL, D_MODEL, 0, 1);
        gemm_wmma<<<g512, blk, 0, stream>>>(xh, wp_l[l][2], bv + l * D_MODEL, nullptr,
                                            nullptr, vtT, ROWS, D_MODEL, D_MODEL, 0, 2);

        attn_wmma<<<agrid, blk, 0, stream>>>(qh, kh, vtT, tau, delta, abh);

        // x = attn @ Wo + bo + x
        gemm_wmma<<<g512, blk, 0, stream>>>(abh, wp_l[l][3], bo + l * D_MODEL, xcur,
                                            xcur, nullptr, ROWS, D_MODEL, D_MODEL, 0, 0);
        // xb = LN1(x), dual f32+f16
        layernorm_k<<<ROWS, 256, 0, stream>>>(xcur, ln1g + l * D_MODEL,
                                              ln1b + l * D_MODEL, xb, xbh);
        // h = gelu(xb @ W1 + b1) -> f16
        gemm_wmma<<<g2048, blk, 0, stream>>>(xbh, wp_l[l][4], b1 + l * D_FF, nullptr,
                                             nullptr, hbh, ROWS, D_FF, D_MODEL, 1, 1);
        // xcur = h @ W2 + b2 + xb ; then x = LN2(xcur) in-place (+ f16 copy)
        gemm_wmma<<<g512, blk, 0, stream>>>(hbh, wp_l[l][5], b2 + l * D_MODEL, xb,
                                            xcur, nullptr, ROWS, D_MODEL, D_FF, 0, 0);
        layernorm_k<<<ROWS, 256, 0, stream>>>(xcur, ln2g + l * D_MODEL,
                                              ln2b + l * D_MODEL, xcur, xh);
    }

    layernorm_k<<<ROWS, 256, 0, stream>>>(xcur, lnfg, lnfb, out, nullptr);
}
